// KernelToeplitzCausalLinear_1898375545201
// MI455X (gfx1250) — matrix operand order repaired and checked by reading, not compiled, not run on hardware
//
#include <hip/hip_runtime.h>
#include <stdint.h>

#define DIM   1024
#define KT    4
#define EB    2048
#define NB    8
#define MTOT  (NB*EB)     // 16384 rows total

#define TM    128         // output rows per block
#define TN    128         // output cols per block
#define XPITCH 40         // ushorts per LDS x row (80B: multiple of 16B, conflict-free)
#define XROWS  132        // 131 rows used (128 + 3 halo)
#define NUNITS (131*8)    // 8B staging units per step
#define WLEN   1280       // reversed weight length incl. zero pad
#define WB_OFF (KT*WLEN)  // second (one-element-shifted) copy

typedef __attribute__((ext_vector_type(16))) __bf16       v16bf;
typedef __attribute__((ext_vector_type(8)))  float        v8f;
typedef __attribute__((ext_vector_type(8)))  unsigned int u32x8;
typedef unsigned long long ull;

static __device__ __forceinline__ unsigned short f2bf(float f) {
  unsigned u = __builtin_bit_cast(unsigned, f);
  unsigned r = (u + 0x7FFFu + ((u >> 16) & 1u)) >> 16;   // round-to-nearest-even
  return (unsigned short)r;
}

// ---- prep: x fp32 -> bf16 in workspace
__global__ __launch_bounds__(256)
void cvt_x_bf16(const float* __restrict__ x, unsigned short* __restrict__ xbf) {
  long i = (long)blockIdx.x * blockDim.x + threadIdx.x;
  const float4 f = ((const float4*)x)[i];
  ushort4 o;
  o.x = f2bf(f.x); o.y = f2bf(f.y); o.z = f2bf(f.z); o.w = f2bf(f.w);
  ((ushort4*)xbf)[i] = o;
}

// ---- main: causal Toeplitz GEMM via bf16 WMMA, fp32 accumulate
__global__ __launch_bounds__(256)
void toeplitz_wmma(const unsigned short* __restrict__ xbf,
                   const float* __restrict__ weight,
                   const float* __restrict__ bias,
                   float* __restrict__ out) {
  __shared__ __align__(16) unsigned short lds_x[2][XROWS * XPITCH]; // double buffer
  __shared__ __align__(16) unsigned short lds_w[2 * KT * WLEN];

  const int tid  = threadIdx.x;
  const int lane = tid & 31;
  // wave coords kept scalar so zero-block skips are SALU branches (EXEC stays all-1s)
  const int wv = __builtin_amdgcn_readfirstlane(tid >> 5);
  const int wm = wv & 3;                     // 32-row M quarter
  const int wn = wv >> 2;                    // 64-col N half
  const int n = lane & 15, half = lane >> 4; // WMMA lane coords (wave32)

  const int  m0 = blockIdx.x * TM;
  const int  t0 = blockIdx.y * TN;
  const long batchBase = (long)(m0 & ~(EB - 1));

  // Stage reversed bf16 weights: wrev[k][p] = w[k][1023-p], 0 for p>1023.
  // Copy at WB_OFF is shifted one element so any (p,p+1) pair is one aligned b32.
  for (int idx = tid; idx < 2 * KT * WLEN; idx += 256) {
    int which = idx / (KT * WLEN);
    int rem   = idx - which * (KT * WLEN);
    int k = rem / WLEN, q = rem - k * WLEN;
    int d = 1023 - (q + which);
    lds_w[idx] = (d >= 0) ? f2bf(weight[k * DIM + d]) : (unsigned short)0;
  }

  v8f acc[2][4] = {};   // 32 (M) x 64 (N) fp32 accumulators per wave

  ull stg[5];           // register staging for next x tile (software pipeline)
  auto issueLoads = [&](int s0v) {
#pragma unroll
    for (int q = 0; q < 5; ++q) {
      int u = tid + q * 256;
      ull v = 0ull;
      if (u < NUNITS) {
        int r = u >> 3, c = u & 7;
        long g = (long)m0 - 3 + r;
        if (g >= batchBase) {
          v = *(const ull*)(xbf + g * DIM + s0v + c * 4);
          __builtin_prefetch(xbf + g * DIM + s0v + 32 + c * 4, 0, 1);
        }
      }
      stg[q] = v;
    }
  };
  auto storeStage = [&](int pb) {
#pragma unroll
    for (int q = 0; q < 5; ++q) {
      int u = tid + q * 256;
      if (u < NUNITS) {
        int r = u >> 3, c = u & 7;
        *(ull*)(&lds_x[pb][r * XPITCH + c * 4]) = stg[q];
      }
    }
  };

  // WMMA body. `boundary` is a literal at each call site; each call site sits
  // in its own loop so accumulator allocation stays stable per loop.
  auto compute = [&](int s0, int pb, bool boundary) {
    if (boundary && s0 > t0 + 64 * wn + 63) return;  // whole wave zero (scalar)
#pragma unroll
    for (int k = 0; k < KT; ++k) {
      // A fragments (16x32 bf16): tap-k == row shift in the haloed tile.
      v16bf a[2];
#pragma unroll
      for (int ms = 0; ms < 2; ++ms) {
        const int r = 32 * wm + 16 * ms + n + k;
        const unsigned char* xb =
            (const unsigned char*)&lds_x[pb][0] + r * (XPITCH * 2) + half * 32;
        const uint4 alo = *(const uint4*)(xb);
        const uint4 ahi = *(const uint4*)(xb + 16);
        u32x8 au;
        au[0] = alo.x; au[1] = alo.y; au[2] = alo.z; au[3] = alo.w;
        au[4] = ahi.x; au[5] = ahi.y; au[6] = ahi.z; au[7] = ahi.w;
        a[ms] = __builtin_bit_cast(v16bf, au);
      }
#pragma unroll
      for (int nt = 0; nt < 4; ++nt) {
        if (!boundary || s0 <= t0 + 64 * wn + 16 * nt + 15) {
          // B fragment (32x16 bf16): B[K][n] = wrev[1023-(t-s)]; the lane's
          // 8 VGPRs are 8 consecutive dwords of the parity-selected copy.
          int base = (t0 + 64 * wn + 16 * nt + n) - s0 - half * 16;
          int p0   = 1023 - base;
          unsigned off = (p0 & 1) ? (unsigned)(WB_OFF + k * WLEN + (p0 - 1))
                                  : (unsigned)(k * WLEN + p0);
          const unsigned* dw = (const unsigned*)(lds_w + off);
          u32x8 bu;
#pragma unroll
          for (int j = 0; j < 8; ++j) bu[j] = dw[j];
          v16bf b = __builtin_bit_cast(v16bf, bu);

          // one B fragment feeds two WMMAs (two M sub-tiles)
          acc[0][nt] = __builtin_amdgcn_wmma_f32_16x16x32_bf16(
              false, a[0], false, b, (short)0, acc[0][nt], false, false);
          acc[1][nt] = __builtin_amdgcn_wmma_f32_16x16x32_bf16(
              false, a[1], false, b, (short)0, acc[1][nt], false, false);
        }
      }
    }
  };

  const int sEnd = t0 + TN;                  // causal: only s-tiles with s <= t_max
  issueLoads(0);
  int pb = 0;
  int s0 = 0;

  // Phase 1 (separate loop): interior steps, every block active -> branchless body
  for (; s0 <= t0; s0 += 32) {
    storeStage(pb);
    __syncthreads();
    if (s0 + 32 < sEnd) issueLoads(s0 + 32); // overlap next tile's loads with WMMA
    compute(s0, pb, false);
    pb ^= 1;
  }
  // Phase 2 (separate loop): 3 boundary steps with scalar zero-block skips
  for (; s0 < sEnd; s0 += 32) {
    storeStage(pb);
    __syncthreads();
    if (s0 + 32 < sEnd) issueLoads(s0 + 32);
    compute(s0, pb, true);
    pb ^= 1;
  }

  // Epilogue: C/D layout -> VGPR v: M = v + half*8, N = lane&15
#pragma unroll
  for (int ms = 0; ms < 2; ++ms) {
    const int mbase = m0 + 32 * wm + 16 * ms + half * 8;
#pragma unroll
    for (int nt = 0; nt < 4; ++nt) {
      int t = t0 + 64 * wn + 16 * nt + n;
      float bv = bias[t];
#pragma unroll
      for (int v = 0; v < 8; ++v)
        out[(long)(mbase + v) * DIM + t] = acc[ms][nt][v] + bv;
    }
  }
}

extern "C" void kernel_launch(void* const* d_in, const int* in_sizes, int n_in,
                              void* d_out, int out_size, void* d_ws, size_t ws_size,
                              hipStream_t stream) {
  const float* x = (const float*)d_in[0];   // (8, 2048, 1024) f32
  const float* w = (const float*)d_in[1];   // (4, 1024) f32
  const float* b = (const float*)d_in[2];   // (1024,) f32
  float* out = (float*)d_out;               // (8, 2048, 1024) f32
  unsigned short* xbf = (unsigned short*)d_ws;  // needs 32 MiB scratch

  long n4 = (long)MTOT * DIM / 4;
  cvt_x_bf16<<<dim3((unsigned)(n4 / 256)), dim3(256), 0, stream>>>(x, xbf);

  dim3 grid(MTOT / TM, DIM / TN);           // 128 x 8 workgroups
  toeplitz_wmma<<<grid, dim3(256), 0, stream>>>(xbf, w, b, out);
}
